// MultiHeadAttention_22187801051539
// MI455X (gfx1250) — matrix operand order repaired and checked by reading, not compiled, not run on hardware
//
#include <hip/hip_runtime.h>
#include <stdint.h>

#define DEVINL __device__ __forceinline__

typedef __attribute__((ext_vector_type(16))) __bf16 v16bf;
typedef __attribute__((ext_vector_type(8)))  float  v8f;
typedef unsigned short u16;

namespace {

constexpr int B  = 8;
constexpr int S  = 2048;
constexpr int H  = 6;
constexpr int DK = 24;
constexpr int DM = 144;
constexpr int BH = B * H;
constexpr int M  = B * S;          // 16384 rows of the [M, DM] activations
constexpr int WPAD = 160;          // K-padded weight row (144 -> 160, zero fill)
constexpr int NQT = S / 32;        // 64 q-tiles (32 rows each) per (b,h)
constexpr int NKT = S / 32;        // 64 key tiles (32 keys each)
constexpr float ATT_SCALE = 0.020412414523193154f;  // 0.1 / sqrt(24)
constexpr float LNEPS = 1e-5f;

union Frag16 {            // 16 bf16 per lane == one WMMA A/B operand slice
    uint4 q[2];
    v16bf v;
};

DEVINL uint32_t pk_bf16(float a, float b) {   // pack two floats -> 2x bf16 (RNE)
#if __has_builtin(__builtin_amdgcn_cvt_pk_bf16_f32)
    auto r = __builtin_amdgcn_cvt_pk_bf16_f32(a, b);
    return __builtin_bit_cast(uint32_t, r);
#else
    uint32_t ua = __builtin_bit_cast(uint32_t, a);
    uint32_t ub = __builtin_bit_cast(uint32_t, b);
    ua = (ua + 0x7FFFu + ((ua >> 16) & 1u)) >> 16;
    ub = (ub + 0x7FFFu + ((ub >> 16) & 1u)) >> 16;
    return ua | (ub << 16);
#endif
}
DEVINL u16 bf16_1(float a) {      // single float -> bf16 (RNE)
    uint32_t ua = __builtin_bit_cast(uint32_t, a);
    return (u16)((ua + 0x7FFFu + ((ua >> 16) & 1u)) >> 16);
}
DEVINL u16 bf16_trunc(float a) {  // truncating bf16: hits d16_hi store path
    return (u16)(__builtin_bit_cast(uint32_t, a) >> 16);
}
DEVINL uint4 pk4(float4 f0, float4 f1) {
    return make_uint4(pk_bf16(f0.x, f0.y), pk_bf16(f0.z, f0.w),
                      pk_bf16(f1.x, f1.y), pk_bf16(f1.z, f1.w));
}

DEVINL float fast_tanh(float x) {
#if __has_builtin(__builtin_amdgcn_tanhf)
    return __builtin_amdgcn_tanhf(x);
#elif __has_builtin(__builtin_amdgcn_tanh_f32)
    return __builtin_amdgcn_tanh_f32(x);
#else
    // tanh(x) = 1 - 2/(e^{2x}+1); saturates cleanly (no inf/inf)
    float e = __expf(2.0f * x);
    return 1.0f - 2.0f * __builtin_amdgcn_rcpf(e + 1.0f);
#endif
}

DEVINL v8f wmma_bf16(v16bf a, v16bf b, v8f c) {
    return __builtin_amdgcn_wmma_f32_16x16x32_bf16(
        /*neg_a=*/false, a, /*neg_b=*/false, b,
        /*c_mod=*/(short)0, c, /*reuse_a=*/false, /*reuse_b=*/false);
}

// A-operand (16x32 bf16) from a bf16 [*,144] row, K chunk kbase in {0,32,64,96,128}.
// A layout: lanes 0-15 row r hold K=kb+{0..7} (v0-3) and kb+{16..23} (v4-7);
// lanes 16-31 hold kb+{8..15} and kb+{24..31}. Chunk 4 upper half is zero.
DEVINL v16bf load_a_bf16(const u16* rowp, int kbase, int lane) {
    int sel = (lane & 16) ? 8 : 0;
    Frag16 fr;
    fr.q[0] = *(const uint4*)(rowp + kbase + sel);
    if (kbase + 16 < DM)
        fr.q[1] = *(const uint4*)(rowp + kbase + 16 + sel);
    else
        fr.q[1] = make_uint4(0, 0, 0, 0);
    return fr.v;
}

// B-operand (32x16 bf16) from bf16 K-padded weights W[144][160] (global or LDS),
// B[k,n] = W[n,k].  Lanes 0-15: col n, K=kb+{0..7,16..23}; lanes 16-31: K=kb+{8..15,24..31}.
DEVINL v16bf load_b_w(const u16* Wp, int ntile, int kbase, int lane) {
    int n  = ntile * 16 + (lane & 15);
    int ko = kbase + ((lane & 16) ? 8 : 0);
    Frag16 fr;
    const uint4* p = (const uint4*)(Wp + (size_t)n * WPAD + ko);
    fr.q[0] = p[0];
    fr.q[1] = p[2];
    return fr.v;
}

// ---------------- pack kernels ----------------

__global__ void __launch_bounds__(256) pack_x_kernel(const float* __restrict__ src,
                                                     u16* __restrict__ dst, int n8) {
    int i = blockIdx.x * blockDim.x + threadIdx.x;
    if (i >= n8) return;
    const float4* p = (const float4*)src + (size_t)i * 2;
    float4 f0 = p[0];
    float4 f1 = p[1];
    *((uint4*)dst + i) = pk4(f0, f1);
}

__global__ void __launch_bounds__(256) pack_w_kernel(const float* __restrict__ Wq,
                                                     const float* __restrict__ Lw,
                                                     u16* __restrict__ Wqp,
                                                     u16* __restrict__ Lwp) {
    int idx = blockIdx.x * blockDim.x + threadIdx.x;
    int total = DM * WPAD;
    if (idx >= 2 * total) return;
    const float* src = (idx < total) ? Wq : Lw;
    u16* dst = (idx < total) ? Wqp : Lwp;
    int e = (idx < total) ? idx : idx - total;
    int n = e / WPAD, k = e % WPAD;
    dst[e] = (k < DM) ? bf16_1(src[n * DM + k]) : (u16)0;
}

__global__ void __launch_bounds__(256) pack_kv_kernel(const float* __restrict__ Kin,
                                                      const float* __restrict__ Vin,
                                                      u16* __restrict__ Kp,
                                                      u16* __restrict__ Vt) {
    int idx = blockIdx.x * blockDim.x + threadIdx.x;
    if (idx >= BH * S) return;
    int s = idx % S;
    size_t bh = (size_t)(idx / S);
    const float* kr = Kin + (bh * S + s) * DK;
    u16* kp = Kp + (bh * S + s) * 32;
#pragma unroll
    for (int d = 0; d < 32; ++d) kp[d] = (d < DK) ? bf16_1(kr[d]) : (u16)0;
    const float* vr = Vin + (bh * S + s) * DK;
#pragma unroll
    for (int d = 0; d < 32; ++d)
        Vt[(bh * 32 + d) * S + s] = (d < DK) ? bf16_1(vr[d]) : (u16)0;
}

// Depth-2 software-pipelined GEMM core: 45 (kc,t) steps, B-fragments from LDS.
DEVINL void gemm_core(const u16* wsm, const v16bf (&af)[5], v8f (&cf)[9], int lane) {
    v16bf bq0 = load_b_w(wsm, 0, 0, lane);
    v16bf bq1 = load_b_w(wsm, 1, 0, lane);
#pragma unroll
    for (int u = 0; u < 45; ++u) {
        int kc = u / 9, t = u % 9;
        v16bf bwork = (u & 1) ? bq1 : bq0;
        if (u + 2 < 45) {
            int un = u + 2;
            v16bf bn = load_b_w(wsm, un % 9, (un / 9) * 32, lane);
            if (u & 1) bq1 = bn; else bq0 = bn;
        }
        cf[t] = wmma_bf16(af[kc], bwork, cf[t]);
    }
#if __has_builtin(__builtin_amdgcn_sched_group_barrier)
    // Pin the pipeline shape the list scheduler keeps undoing:
    //   6 DS reads (3 frags in flight), then 42 x (1 WMMA, 2 DS reads), 3 WMMAs.
    // Masks: 0x008 = MFMA/WMMA, 0x100 = DS read.
    __builtin_amdgcn_sched_group_barrier(0x100, 6, 0);
#pragma unroll
    for (int u = 0; u < 42; ++u) {
        __builtin_amdgcn_sched_group_barrier(0x008, 1, 0);
        __builtin_amdgcn_sched_group_barrier(0x100, 2, 0);
    }
#pragma unroll
    for (int u = 0; u < 3; ++u)
        __builtin_amdgcn_sched_group_barrier(0x008, 1, 0);
#endif
}

// ---------------- q projection: q = Q @ Wq^T + b  -> bf16 [M,144] ----------------
// Weights staged once per block into LDS; B-fragments then come from LDS.

__global__ void __launch_bounds__(256) qproj_kernel(const u16* __restrict__ Qbf,
                                                    const u16* __restrict__ Wqp,
                                                    const float* __restrict__ Wqb,
                                                    u16* __restrict__ qb) {
    __shared__ __align__(16) u16 wsm[DM * WPAD];   // 45 KB
    int tid  = threadIdx.x;
    for (int i = tid; i < DM * WPAD / 8; i += 256)
        ((uint4*)wsm)[i] = ((const uint4*)Wqp)[i];
    __syncthreads();

    int wid  = (blockIdx.x * blockDim.x + tid) >> 5;   // one wave = 16 rows
    int lane = tid & 31;
    int row0 = wid * 16;
    const u16* arow = Qbf + (size_t)(row0 + (lane & 15)) * DM;

    v16bf af[5];
#pragma unroll
    for (int kc = 0; kc < 5; ++kc) af[kc] = load_a_bf16(arow, kc * 32, lane);

    v8f cf[9];
#pragma unroll
    for (int t = 0; t < 9; ++t) cf[t] = (v8f){0, 0, 0, 0, 0, 0, 0, 0};

    gemm_core(wsm, af, cf, lane);

    int col = lane & 15, roff = (lane & 16) ? 8 : 0;
#pragma unroll
    for (int t = 0; t < 9; ++t) {
        int n = t * 16 + col;
        float bias = Wqb[n];
#pragma unroll
        for (int i = 0; i < 8; ++i) {
            int r = row0 + i + roff;
            qb[(size_t)r * DM + n] = bf16_1(cf[t][i] + bias);
        }
    }
}

// ---------------- fused attention: tanh(scale * q K^T) @ V -> bf16 ctx ----------------
// Block = 8 waves on the same (b,h); 32 q rows per wave.  K/V tiles flow through
// a 3-stage pipeline: global -> registers (kt+2) -> LDS (kt+1) -> compute (kt).

__global__ void __launch_bounds__(256) attn_kernel(const u16* __restrict__ qb,
                                                   const u16* __restrict__ Kp,
                                                   const u16* __restrict__ Vt,
                                                   u16* __restrict__ ctxb) {
    __shared__ __align__(16) u16 ksm[2][32 * 32];  // 2 x 2KB K tiles
    __shared__ __align__(16) u16 vsm[2][32 * 32];  // 2 x 2KB V tiles
    __shared__ __align__(16) u16 smA[8 * 32 * 40]; // per-wave attn scratch (20KB)

    int tid  = threadIdx.x;
    int wloc = tid >> 5;
    int lane = tid & 31;
    int bh   = blockIdx.x >> 3;                 // 8 blocks per (b,h)
    int qt   = ((blockIdx.x & 7) << 3) + wloc;  // this wave's 32-row q tile
    int b = bh / H, h = bh % H;
    u16* smw = smA + wloc * (32 * 40);

    int sel8 = (lane & 16) ? 8 : 0;
    int col = lane & 15, roff = (lane & 16) ? 8 : 0;
    int ko = (lane & 16) ? 8 : 0;

    const u16* kb_p = Kp + (size_t)bh * S * 32;
    const u16* vb_p = Vt + (size_t)bh * 32 * S;

    // each thread owns one uint4 of the 4KB (K+V) tile pair
    int t2 = tid & 127;
    int vdim = t2 >> 2, vpart = t2 & 3;
    auto fetch = [&](int kt) -> uint4 {
        if (tid < 128)
            return ((const uint4*)(kb_p + (size_t)kt * 32 * 32))[tid];
        return ((const uint4*)(vb_p + (size_t)vdim * S + kt * 32))[vpart];
    };
    auto put = [&](int bi, uint4 v) {
        if (tid < 128)
            ((uint4*)ksm[bi])[tid] = v;
        else
            ((uint4*)(vsm[bi] + vdim * 32))[vpart] = v;
    };

    // q A-fragments (2 x 16x32, K=d_k padded 24->32)
    Frag16 aq[2];
#pragma unroll
    for (int j = 0; j < 2; ++j) {
        const u16* qrow =
            qb + (size_t)(b * S + qt * 32 + j * 16 + (lane & 15)) * DM + h * DK;
        aq[j].q[0] = *(const uint4*)(qrow + sel8);
        uint4 ahi = *(const uint4*)(qrow + 16);      // K=16..23 (real for lanes<16)
        if (lane & 16) ahi = make_uint4(0, 0, 0, 0); // K=24..31 pad
        aq[j].q[1] = ahi;
    }

    v8f c[2][2];
#pragma unroll
    for (int j = 0; j < 2; ++j)
#pragma unroll
        for (int hn = 0; hn < 2; ++hn) c[j][hn] = (v8f){0, 0, 0, 0, 0, 0, 0, 0};
    const v8f vz = (v8f){0, 0, 0, 0, 0, 0, 0, 0};

    put(0, fetch(0));          // tile 0 -> LDS (one-time exposed latency)
    uint4 sreg = fetch(1);     // tile 1 -> registers
    __syncthreads();

    for (int kt = 0; kt < NKT; ++kt) {
        int cur = kt & 1;

        // K fragments from the block-shared LDS tile
        Frag16 bk[2];
#pragma unroll
        for (int hn = 0; hn < 2; ++hn) {
            const u16* kp_ = ksm[cur] + (hn * 16 + (lane & 15)) * 32 + ko;
            bk[hn].q[0] = *(const uint4*)(kp_);
            bk[hn].q[1] = *(const uint4*)(kp_ + 16);
        }

        v8f s[2][2];
#pragma unroll
        for (int j = 0; j < 2; ++j) {
            s[j][0] = wmma_bf16(aq[j].v, bk[0].v, vz);
            s[j][1] = wmma_bf16(aq[j].v, bk[1].v, vz);
        }

        // V fragments (issue before tanh so DS latency hides under VALU)
        Frag16 bv[2];
#pragma unroll
        for (int hn = 0; hn < 2; ++hn) {
            const u16* vp_ = vsm[cur] + (hn * 16 + (lane & 15)) * 32 + ko;
            bv[hn].q[0] = *(const uint4*)(vp_);
            bv[hn].q[1] = *(const uint4*)(vp_ + 16);
        }

        // tanh activation; spill to wave-private LDS as bf16 (high-half stores)
#pragma unroll
        for (int j = 0; j < 2; ++j) {
            u16* smt = smw + j * (16 * 40);
#pragma unroll
            for (int i = 0; i < 8; ++i) {
                smt[(i + roff) * 40 + col]      = bf16_trunc(fast_tanh(ATT_SCALE * s[j][0][i]));
                smt[(i + roff) * 40 + 16 + col] = bf16_trunc(fast_tanh(ATT_SCALE * s[j][1][i]));
            }
        }
        asm volatile("s_wait_dscnt 0" ::: "memory");  // wave-private tile: no barrier

        // re-read as A-fragments (16 rows x 32 keys each)
        Frag16 aa[2];
#pragma unroll
        for (int j = 0; j < 2; ++j) {
            const u16* ap = smw + j * (16 * 40) + (lane & 15) * 40 + sel8;
            aa[j].q[0] = *(const uint4*)(ap);
            aa[j].q[1] = *(const uint4*)(ap + 16);
        }

#pragma unroll
        for (int j = 0; j < 2; ++j) {
            c[j][0] = wmma_bf16(aa[j].v, bv[0].v, c[j][0]);
            c[j][1] = wmma_bf16(aa[j].v, bv[1].v, c[j][1]);
        }

        // drain pipeline: registers -> LDS for kt+1, then prefetch kt+2
        if (kt + 1 < NKT) {
            put(cur ^ 1, sreg);
            if (kt + 2 < NKT) sreg = fetch(kt + 2);
        }
        __syncthreads();   // publish tile kt+1; retire reads of recycled buffer
    }

    // store bf16 context [M,144] (head h occupies cols h*24 .. h*24+23)
#pragma unroll
    for (int j = 0; j < 2; ++j)
#pragma unroll
        for (int i = 0; i < 8; ++i) {
            int r = b * S + qt * 32 + j * 16 + i + roff;
            u16* crow = ctxb + (size_t)r * DM + h * DK;
            crow[col] = bf16_1(c[j][0][i]);
            if (col < 8) crow[16 + col] = bf16_1(c[j][1][i]);
        }
}

// ---------------- output GEMM + bias + residual + LayerNorm ----------------

__global__ void __launch_bounds__(256) out_ln_kernel(const u16* __restrict__ ctxb,
                                                     const u16* __restrict__ Lwp,
                                                     const float* __restrict__ lb,
                                                     const float* __restrict__ Qres,
                                                     const float* __restrict__ g,
                                                     const float* __restrict__ bb,
                                                     float* __restrict__ out) {
    __shared__ __align__(16) u16 wsm[DM * WPAD];   // 45 KB
    int tid  = threadIdx.x;
    for (int i = tid; i < DM * WPAD / 8; i += 256)
        ((uint4*)wsm)[i] = ((const uint4*)Lwp)[i];
    __syncthreads();

    int wid  = (blockIdx.x * blockDim.x + tid) >> 5;
    int lane = tid & 31;
    int row0 = wid * 16;
    const u16* arow = ctxb + (size_t)(row0 + (lane & 15)) * DM;

    v16bf af[5];
#pragma unroll
    for (int kc = 0; kc < 5; ++kc) af[kc] = load_a_bf16(arow, kc * 32, lane);

    v8f cf[9];
#pragma unroll
    for (int t = 0; t < 9; ++t) cf[t] = (v8f){0, 0, 0, 0, 0, 0, 0, 0};

    gemm_core(wsm, af, cf, lane);

    int col = lane & 15, roff = (lane & 16) ? 8 : 0;
#pragma unroll
    for (int t = 0; t < 9; ++t) {
        int n = t * 16 + col;
        float bias = lb[n];
#pragma unroll
        for (int i = 0; i < 8; ++i)
            cf[t][i] += bias + Qres[(size_t)(row0 + i + roff) * DM + n];
    }

    // LayerNorm: each output row lives across one 16-lane group -> xor reduce 1,2,4,8
    float mean[8], rstd[8];
#pragma unroll
    for (int i = 0; i < 8; ++i) {
        float s = 0.f, s2 = 0.f;
#pragma unroll
        for (int t = 0; t < 9; ++t) { float x = cf[t][i]; s += x; s2 += x * x; }
        s  += __shfl_xor(s, 1);  s2 += __shfl_xor(s2, 1);
        s  += __shfl_xor(s, 2);  s2 += __shfl_xor(s2, 2);
        s  += __shfl_xor(s, 4);  s2 += __shfl_xor(s2, 4);
        s  += __shfl_xor(s, 8);  s2 += __shfl_xor(s2, 8);
        float mu  = s * (1.0f / DM);
        float var = s2 * (1.0f / DM) - mu * mu;
        mean[i] = mu;
        rstd[i] = rsqrtf(var + LNEPS);
    }

#pragma unroll
    for (int t = 0; t < 9; ++t) {
        int n = t * 16 + col;
        float gg = g[n], bv = bb[n];
#pragma unroll
        for (int i = 0; i < 8; ++i)
            out[(size_t)(row0 + i + roff) * DM + n] =
                (cf[t][i] - mean[i]) * rstd[i] * gg + bv;
    }
}

} // namespace

extern "C" void kernel_launch(void* const* d_in, const int* in_sizes, int n_in,
                              void* d_out, int out_size, void* d_ws, size_t ws_size,
                              hipStream_t stream) {
    (void)in_sizes; (void)n_in; (void)out_size; (void)ws_size;
    const float* Q    = (const float*)d_in[0];
    const float* Kin  = (const float*)d_in[1];
    const float* Vin  = (const float*)d_in[2];
    // d_in[3] = attn_mask : faithfully ignored (reference drops masked_fill result)
    const float* Wq_w = (const float*)d_in[4];
    const float* Wq_b = (const float*)d_in[5];
    const float* Lw   = (const float*)d_in[6];
    const float* Lb   = (const float*)d_in[7];
    const float* Lng  = (const float*)d_in[8];
    const float* Lnb  = (const float*)d_in[9];
    float* out = (float*)d_out;

    char* ws = (char*)d_ws;
    size_t off = 0;
    auto carve = [&](size_t bytes) -> void* {
        void* p = ws + off;
        off += (bytes + 255) & ~(size_t)255;
        return p;
    };
    u16*  Qbf = (u16*)carve((size_t)M * DM * sizeof(u16));        // bf16 input Q
    u16*  qb  = (u16*)carve((size_t)M * DM * sizeof(u16));        // bf16 q proj
    u16*  Kp  = (u16*)carve((size_t)BH * S * 32 * sizeof(u16));   // bf16 K, d_k pad 32
    u16*  Vt  = (u16*)carve((size_t)BH * 32 * S * sizeof(u16));   // bf16 V^T, pad 32
    u16*  Wqp = (u16*)carve((size_t)DM * WPAD * sizeof(u16));
    u16*  Lwp = (u16*)carve((size_t)DM * WPAD * sizeof(u16));
    u16*  ctxb = (u16*)carve((size_t)M * DM * sizeof(u16));       // bf16 context

    hipLaunchKernelGGL(pack_x_kernel, dim3((M * DM / 8 + 255) / 256), dim3(256),
                       0, stream, Q, Qbf, M * DM / 8);
    hipLaunchKernelGGL(pack_w_kernel, dim3((2 * DM * WPAD + 255) / 256), dim3(256),
                       0, stream, Wq_w, Lw, Wqp, Lwp);
    hipLaunchKernelGGL(pack_kv_kernel, dim3((BH * S + 255) / 256), dim3(256),
                       0, stream, Kin, Vin, Kp, Vt);
    hipLaunchKernelGGL(qproj_kernel, dim3((M / 16) / 8), dim3(256),
                       0, stream, Qbf, Wqp, Wq_b, qb);
    hipLaunchKernelGGL(attn_kernel, dim3((BH * NQT) / 8), dim3(256),
                       0, stream, qb, Kp, Vt, ctxb);
    hipLaunchKernelGGL(out_ln_kernel, dim3((M / 16) / 8), dim3(256),
                       0, stream, ctxb, Lwp, Lb, Q, Lng, Lnb, out);
}
